// Kernel_65481071404369
// MI455X (gfx1250) — matrix operand order repaired and checked by reading, not compiled, not run on hardware
//
#include <hip/hip_runtime.h>

typedef __attribute__((ext_vector_type(16))) __bf16 v16bf;
typedef __attribute__((ext_vector_type(8)))  __bf16 v8bf;
typedef __attribute__((ext_vector_type(8)))  float  v8f;
typedef __attribute__((ext_vector_type(4)))  float  v4f;

#define TILE 128
#define KSTEP 32
#define ROW_BF16 40                   // padded row stride: 80 B (16B aligned, 20-dword bank spread)
#define TILE_BF16 (TILE * ROW_BF16)   // 5120 bf16 = 10240 B per tile
#define NT 4                          // tiles per buffer: xhi, xlo, yhi, ylo
#define SMEM_BYTES (2 * NT * TILE_BF16 * 2)  // double buffered = 81920 B

// ---------------------------------------------------------------------------
// Async global->LDS b128 copy (CDNA5, tracked by ASYNCcnt).
// ---------------------------------------------------------------------------
__device__ __forceinline__ void async_b128(unsigned lds_byte, const __bf16* g) {
    asm volatile("global_load_async_to_lds_b128 %0, %1, off"
                 :: "v"(lds_byte), "v"(g)
                 : "memory");
}
__device__ __forceinline__ void wait_async0() {
    asm volatile("s_wait_asynccnt 0x0" ::: "memory");
}

// ---------------------------------------------------------------------------
// Pre-pass 1: row squared norms.
// ---------------------------------------------------------------------------
__global__ __launch_bounds__(256) void rowsq_kernel(const float* __restrict__ x,
                                                    const float* __restrict__ y,
                                                    float* __restrict__ x2,
                                                    float* __restrict__ y2,
                                                    int N, int M, int D) {
    int idx = blockIdx.x * blockDim.x + threadIdx.x;
    const float* src; float* dst; int row;
    if (idx < N)          { src = x; dst = x2; row = idx; }
    else if (idx < N + M) { src = y; dst = y2; row = idx - N; }
    else return;
    const v4f* p = (const v4f*)(src + (size_t)row * D);
    float s = 0.f;
#pragma unroll 4
    for (int i = 0; i < D / 4; ++i) {
        v4f v = p[i];
        s += v.x * v.x + v.y * v.y + v.z * v.z + v.w * v.w;
    }
    dst[row] = s;
}

// ---------------------------------------------------------------------------
// Pre-pass 2: one-shot f32 -> (hi,lo) bf16 split. hi = rne(v), lo = rne(v - hi).
// Done once per element instead of once per tile => removes VALU from hot loop.
// ---------------------------------------------------------------------------
__global__ __launch_bounds__(256) void split_kernel(const float* __restrict__ x,
                                                    const float* __restrict__ y,
                                                    __bf16* __restrict__ xhi, __bf16* __restrict__ xlo,
                                                    __bf16* __restrict__ yhi, __bf16* __restrict__ ylo,
                                                    int nx8, int ny8) {
    int idx = blockIdx.x * blockDim.x + threadIdx.x;
    const float* src; __bf16 *hid, *lod; int g;
    if (idx < nx8)            { src = x; hid = xhi; lod = xlo; g = idx; }
    else if (idx < nx8 + ny8) { src = y; hid = yhi; lod = ylo; g = idx - nx8; }
    else return;
    size_t base = (size_t)g * 8;
    v4f a = *(const v4f*)(src + base);
    v4f b = *(const v4f*)(src + base + 4);
    float vals[8] = {a.x, a.y, a.z, a.w, b.x, b.y, b.z, b.w};
    v8bf hv, lv;
#pragma unroll
    for (int e = 0; e < 8; ++e) {
        __bf16 hbf = (__bf16)vals[e];
        hv[e] = hbf;
        lv[e] = (__bf16)(vals[e] - (float)hbf);
    }
    *(v8bf*)(hid + base) = hv;
    *(v8bf*)(lod + base) = lv;
}

// ---------------------------------------------------------------------------
// Main kernel. Block = 256 threads (8 waves) -> 128x128 dist2 tile.
// Wave = 32 rows x 64 cols = 2x4 accumulators of 16x16.
// K loop: double-buffered async-to-LDS staging of bf16 hi/lo tiles,
// ds_load_b128 fragment builds, 3-pass split-bf16 WMMA accumulation.
// ---------------------------------------------------------------------------
__global__ __launch_bounds__(256) void rbf_wmma_kernel(
        const __bf16* __restrict__ xhi, const __bf16* __restrict__ xlo,
        const __bf16* __restrict__ yhi, const __bf16* __restrict__ ylo,
        const float* __restrict__ gamma,
        const float* __restrict__ x2, const float* __restrict__ y2,
        float* __restrict__ out, int N, int M, int D) {
    extern __shared__ __bf16 smem[];
    const unsigned smem_base = (unsigned)(uintptr_t)smem;  // LDS byte offset (low 32b of shared aperture addr)

    const int tid  = threadIdx.x;
    const int lane = tid & 31;
    const int wave = tid >> 5;
    const int lrow = lane & 15;
    const int h    = lane >> 4;

    const int block_row = blockIdx.y * TILE;  // x rows
    const int block_col = blockIdx.x * TILE;  // y rows

    const int wrow = (wave >> 1) * 32;
    const int wcol = (wave & 1) * 64;

    const __bf16* srcs[NT] = {xhi, xlo, yhi, ylo};

    v8f acc[2][4];
#pragma unroll
    for (int i = 0; i < 2; ++i)
#pragma unroll
        for (int j = 0; j < 4; ++j) { v8f z = {}; acc[i][j] = z; }

    // ---- stage one K-step (4 tiles x 512 b128) into buffer `buf` ----
    auto stage = [&](int buf, int kk) {
#pragma unroll
        for (int t = 0; t < NT; ++t) {
            const int row_base = (t < 2) ? block_row : block_col;
#pragma unroll
            for (int i = 0; i < 2; ++i) {
                int flat = tid + i * 256;   // b128 index within tile (512 total)
                int r = flat >> 2;          // 4 b128 per 32-bf16 row
                int q = flat & 3;
                const __bf16* g = srcs[t] + (size_t)(row_base + r) * D + kk + q * 8;
                unsigned lds = smem_base +
                               (unsigned)(((buf * NT + t) * TILE_BF16 + r * ROW_BF16) * 2 + q * 16);
                async_b128(lds, g);
            }
        }
    };

    stage(0, 0);

    const int nsteps = D / KSTEP;
    for (int step = 0; step < nsteps; ++step) {
        const int cur = step & 1;
        wait_async0();
        __syncthreads();
        if (step + 1 < nsteps) stage(cur ^ 1, (step + 1) * KSTEP);

        const __bf16* bufp = smem + (size_t)cur * NT * TILE_BF16;

        // A fragments: 16-bit A 16x32 layout, K runs [8h..8h+7] and [16+8h..23+8h]
        v16bf a_hi[2], a_lo[2];
#pragma unroll
        for (int rt = 0; rt < 2; ++rt) {
            const __bf16* ph = bufp + (wrow + rt * 16 + lrow) * ROW_BF16;
            v8bf h0 = *(const v8bf*)(ph + 8 * h);
            v8bf h1 = *(const v8bf*)(ph + 16 + 8 * h);
            a_hi[rt] = __builtin_shufflevector(h0, h1, 0,1,2,3,4,5,6,7,8,9,10,11,12,13,14,15);
            const __bf16* pl = ph + TILE_BF16;
            v8bf l0 = *(const v8bf*)(pl + 8 * h);
            v8bf l1 = *(const v8bf*)(pl + 16 + 8 * h);
            a_lo[rt] = __builtin_shufflevector(l0, l1, 0,1,2,3,4,5,6,7,8,9,10,11,12,13,14,15);
        }

        // B fragments: 16-bit B 32x16 layout, K runs [8h..8h+7] and [16+8h..23+8h]
#pragma unroll
        for (int ct = 0; ct < 4; ++ct) {
            const __bf16* pbh = bufp + 2 * TILE_BF16 + (wcol + ct * 16 + lrow) * ROW_BF16 + 8 * h;
            v8bf b0 = *(const v8bf*)pbh;
            v8bf b1 = *(const v8bf*)(pbh + 16);
            v16bf b_hi = __builtin_shufflevector(b0, b1, 0,1,2,3,4,5,6,7,8,9,10,11,12,13,14,15);
            const __bf16* pbl = pbh + TILE_BF16;
            v8bf c0 = *(const v8bf*)pbl;
            v8bf c1 = *(const v8bf*)(pbl + 16);
            v16bf b_lo = __builtin_shufflevector(c0, c1, 0,1,2,3,4,5,6,7,8,9,10,11,12,13,14,15);
#pragma unroll
            for (int rt = 0; rt < 2; ++rt) {
                acc[rt][ct] = __builtin_amdgcn_wmma_f32_16x16x32_bf16(
                    false, a_hi[rt], false, b_hi, (short)0, acc[rt][ct], false, false);
                acc[rt][ct] = __builtin_amdgcn_wmma_f32_16x16x32_bf16(
                    false, a_hi[rt], false, b_lo, (short)0, acc[rt][ct], false, false);
                acc[rt][ct] = __builtin_amdgcn_wmma_f32_16x16x32_bf16(
                    false, a_lo[rt], false, b_hi, (short)0, acc[rt][ct], false, false);
            }
        }
        __syncthreads();
    }

    // ---- epilogue: dist2 -> exp(-gamma*dist2); coalesced float4 (G=4) stores ----
    v4f g = *(const v4f*)gamma;
#pragma unroll
    for (int rt = 0; rt < 2; ++rt) {
        const int rbase = block_row + wrow + rt * 16 + 8 * h;   // C/D: m = 8h + e
#pragma unroll
        for (int ct = 0; ct < 4; ++ct) {
            const int col = block_col + wcol + ct * 16 + lrow;  // C/D: n = lane&15
            const float ysq = y2[col];
#pragma unroll
            for (int e = 0; e < 8; ++e) {
                const int row = rbase + e;
                float d = x2[row] + ysq - 2.0f * acc[rt][ct][e];
                d = fmaxf(d, 0.0f);
                v4f o;
                o.x = __expf(-g.x * d);
                o.y = __expf(-g.y * d);
                o.z = __expf(-g.z * d);
                o.w = __expf(-g.w * d);
                *(v4f*)(out + ((size_t)row * M + col) * 4) = o;
            }
        }
    }
}

extern "C" void kernel_launch(void* const* d_in, const int* in_sizes, int n_in,
                              void* d_out, int out_size, void* d_ws, size_t ws_size,
                              hipStream_t stream) {
    const float* x     = (const float*)d_in[0];
    const float* y     = (const float*)d_in[1];
    const float* gamma = (const float*)d_in[2];
    float* out = (float*)d_out;

    const int D = 256;
    const int N = in_sizes[0] / D;   // 4096
    const int M = in_sizes[1] / D;   // 4096

    // Workspace layout: x2[N] f32 | y2[M] f32 | xhi | xlo | yhi | ylo (bf16, N*D/M*D each)
    float* x2 = (float*)d_ws;
    float* y2 = x2 + N;
    __bf16* xhi = (__bf16*)((char*)d_ws + (size_t)(N + M) * sizeof(float));
    __bf16* xlo = xhi + (size_t)N * D;
    __bf16* yhi = xlo + (size_t)N * D;
    __bf16* ylo = yhi + (size_t)M * D;

    rowsq_kernel<<<(N + M + 255) / 256, 256, 0, stream>>>(x, y, x2, y2, N, M, D);

    int nx8 = N * D / 8, ny8 = M * D / 8;
    split_kernel<<<(nx8 + ny8 + 255) / 256, 256, 0, stream>>>(x, y, xhi, xlo, yhi, ylo, nx8, ny8);

    dim3 grid(M / TILE, N / TILE);
    rbf_wmma_kernel<<<grid, 256, SMEM_BYTES, stream>>>(xhi, xlo, yhi, ylo, gamma, x2, y2,
                                                       out, N, M, D);
}
